// RPEAttention_4733053960459
// MI455X (gfx1250) — matrix operand order repaired and checked by reading, not compiled, not run on hardware
//
#include <hip/hip_runtime.h>
#include <hip/hip_bf16.h>
#include <stdint.h>

typedef __attribute__((ext_vector_type(16))) __bf16 v16bf;
typedef __attribute__((ext_vector_type(8)))  float  v8f;

struct Frag { union { v16bf v; uint32_t u[8]; }; };

__device__ __forceinline__ uint32_t pack2bf(float a, float b) {
  union { __bf16 h[2]; uint32_t u; } t;
  t.h[0] = (__bf16)a; t.h[1] = (__bf16)b;
  return t.u;
}

// A-fragment (16x32 bf16, MxK): pair j (j=0..7) covers K = k0, k0+1 for lane L
__device__ __forceinline__ int a_k0(int j, int lane) {
  return ((j < 4) ? 0 : 16) + ((lane < 16) ? 0 : 8) + 2 * (j & 3);
}
// B-fragment (32x16 bf16, KxN): pair j covers K = k0, k0+1; N = lane%16
__device__ __forceinline__ int b_k0(int j, int lane) {
  return ((j < 4) ? 0 : 16) + ((lane < 16) ? 0 : 8) + 2 * (j & 3);
}

__device__ __forceinline__ v8f wmma_bf16(const Frag& A, const Frag& B, v8f c) {
  return __builtin_amdgcn_wmma_f32_16x16x32_bf16(false, A.v, false, B.v,
                                                 (short)0, c, false, false);
}

// ---------------- converts ----------------
__global__ void k_cvt_bf16(const float* __restrict__ s, __bf16* __restrict__ d,
                           int n, float scl) {
  int i = blockIdx.x * blockDim.x + threadIdx.x;
  if (i < n) d[i] = (__bf16)(s[i] * scl);
}

// ---------------- qkv GEMM: [4096,512] x [512,1536]^T -> q,k,v bf16 ----------------
// x: [B=4][C=512][N=1024] f32.  wq: [1536][512] bf16 row-major.
// q: [bh][tok][64] bf16 (pre-scaled 0.125); k: same unscaled; v: [bh][d][tok] bf16.
__global__ void k_qkv(const float* __restrict__ x, const __bf16* __restrict__ wq,
                      __bf16* __restrict__ qb, __bf16* __restrict__ kb,
                      __bf16* __restrict__ vb) {
  const int lane = threadIdx.x & 31;
  const int ntile = blockIdx.x;   // 0..95  (oc tiles)
  const int mtile = blockIdx.y;   // 0..255 (token tiles)
  const uint32_t* w32 = (const uint32_t*)wq;
  const int mrow = mtile * 16 + (lane & 15);
  const int ab = mrow >> 10, antok = mrow & 1023;
  const int oc_b = ntile * 16 + (lane & 15);
  v8f c = {};
  for (int kc = 0; kc < 16; ++kc) {
    Frag A, B;
#pragma unroll
    for (int j = 0; j < 8; ++j) {
      int k0 = kc * 32 + a_k0(j, lane);
      A.u[j] = pack2bf(x[((size_t)ab * 512 + k0) * 1024 + antok],
                       x[((size_t)ab * 512 + k0 + 1) * 1024 + antok]);
      B.u[j] = w32[((size_t)oc_b * 512 + kc * 32 + b_k0(j, lane)) >> 1];
    }
    c = wmma_bf16(A, B, c);
  }
  // part / head / batch are uniform across the whole tile -> one scalar branch,
  // straight-line per-element stores (no per-element divergence).
  const int part = ntile >> 5;            // (ntile*16)/512
  const int hh   = (ntile >> 2) & 7;      // ((ntile*16)/64)%8
  const int d    = ((ntile & 3) << 4) + (lane & 15);
  const int btk  = mtile * 16;            // first token of tile (batch-uniform)
  const int b    = btk >> 10;
  const size_t bh = (size_t)(b * 8 + hh);
  if (part == 0) {
#pragma unroll
    for (int r = 0; r < 8; ++r) {
      int ntok = (btk & 1023) + r + ((lane < 16) ? 0 : 8);
      qb[(bh * 1024 + ntok) * 64 + d] = (__bf16)(c[r] * 0.125f);
    }
  } else if (part == 1) {
#pragma unroll
    for (int r = 0; r < 8; ++r) {
      int ntok = (btk & 1023) + r + ((lane < 16) ? 0 : 8);
      kb[(bh * 1024 + ntok) * 64 + d] = (__bf16)c[r];
    }
  } else {
#pragma unroll
    for (int r = 0; r < 8; ++r) {
      int ntok = (btk & 1023) + r + ((lane < 16) ? 0 : 8);
      vb[(bh * 64 + d) * 1024 + ntok] = (__bf16)c[r];
    }
  }
}

// ---------------- rpe bias GEMM: out[bh][row][240] = a[bh][row][:] . tbl[h][bucket][:] ----
__global__ void k_rpe_gemm(const __bf16* __restrict__ a, const __bf16* __restrict__ tbl,
                           float* __restrict__ out) {
  const int lane = threadIdx.x & 31;
  const int bh = blockIdx.z, h = bh & 7;
  const int btile = blockIdx.x;   // 0..14 (bucket tiles, 240 padded cols)
  const int rtile = blockIdx.y;   // 0..63
  const uint32_t* a32 = (const uint32_t*)a;
  const uint32_t* t32 = (const uint32_t*)tbl;
  const int arow = rtile * 16 + (lane & 15);
  const int bucket = btile * 16 + (lane & 15);
  v8f c = {};
  for (int kc = 0; kc < 2; ++kc) {
    Frag A, B;
#pragma unroll
    for (int j = 0; j < 8; ++j) {
      A.u[j] = a32[(((size_t)bh * 1024 + arow) * 64 + kc * 32 + a_k0(j, lane)) >> 1];
      B.u[j] = (bucket < 225)
                 ? t32[(((size_t)h * 225 + bucket) * 64 + kc * 32 + b_k0(j, lane)) >> 1]
                 : 0u;
    }
    c = wmma_bf16(A, B, c);
  }
#pragma unroll
  for (int r = 0; r < 8; ++r) {
    int m = r + ((lane < 16) ? 0 : 8);
    out[((size_t)bh * 1024 + rtile * 16 + m) * 240 + btile * 16 + (lane & 15)] = c[r];
  }
}

// ---------------- fused attention: QK^T + iRPE biases + softmax + bucket-sum + P.V ----
__global__ void __launch_bounds__(128) k_attn(
    const __bf16* __restrict__ qb, const __bf16* __restrict__ kb,
    const __bf16* __restrict__ vb, const float* __restrict__ tq,
    const float* __restrict__ tk, const float* __restrict__ rpe_v,
    __bf16* __restrict__ obf) {
  __shared__ float sP[16][1032];   // full score/prob rows for 16 queries
  __shared__ float sB[16][226];    // per-row bucket sums
  const int tid = threadIdx.x;
  const int lane = tid & 31;
  const int wv = tid >> 5;
  const int qtile = blockIdx.x;    // 0..63
  const int bh = blockIdx.y;       // 0..31
  const int b = bh >> 3, h = bh & 7;
  const uint32_t* q32 = (const uint32_t*)qb;
  const uint32_t* k32 = (const uint32_t*)kb;
  const uint32_t* v32 = (const uint32_t*)vb;

  for (int idx = tid; idx < 16 * 226; idx += 128) (&sB[0][0])[idx] = 0.f;

  // q A-fragments (reused across all key tiles)
  Frag aq[2];
  {
    int row = qtile * 16 + (lane & 15);
    for (int kc = 0; kc < 2; ++kc)
#pragma unroll
      for (int j = 0; j < 8; ++j)
        aq[kc].u[j] = q32[(((size_t)bh * 1024 + row) * 64 + kc * 32 + a_k0(j, lane)) >> 1];
  }

  // Phase 1: S = q.k^T + tq-gather + tk-gather
  for (int jt = wv; jt < 64; jt += 4) {
    v8f c = {};
    for (int kc = 0; kc < 2; ++kc) {
      Frag B;
#pragma unroll
      for (int j = 0; j < 8; ++j) {
        int key = jt * 16 + (lane & 15);
        B.u[j] = k32[(((size_t)bh * 1024 + key) * 64 + kc * 32 + b_k0(j, lane)) >> 1];
      }
      c = wmma_bf16(aq[kc], B, c);
    }
#pragma unroll
    for (int r = 0; r < 8; ++r) {
      int m = r + ((lane < 16) ? 0 : 8);
      int i = qtile * 16 + m;
      int jj = jt * 16 + (lane & 15);
      int yi = i >> 5, xi = i & 31, yj = jj >> 5, xj = jj & 31;
      int dy = min(max(yi - yj, -7), 7) + 7;
      int dx = min(max(xi - xj, -7), 7) + 7;
      int bij = dy * 15 + dx;
      int bji = (14 - dy) * 15 + (14 - dx);
      sP[m][jj] = c[r] + tq[((size_t)bh * 1024 + i) * 240 + bij]
                       + tk[((size_t)bh * 1024 + jj) * 240 + bji];
    }
  }
  __syncthreads();

  // Phase 2: softmax (8 lanes per row) + bucket-sum via LDS atomics
  {
    int r = tid >> 3, sub = tid & 7;
    int i = qtile * 16 + r;
    float mx = -1e30f;
    for (int j = sub; j < 1024; j += 8) mx = fmaxf(mx, sP[r][j]);
    for (int o = 4; o; o >>= 1) mx = fmaxf(mx, __shfl_xor(mx, o, 8));
    float sm = 0.f;
    for (int j = sub; j < 1024; j += 8) {
      float e = __expf(sP[r][j] - mx); sP[r][j] = e; sm += e;
    }
    for (int o = 4; o; o >>= 1) sm += __shfl_xor(sm, o, 8);
    float inv = 1.f / sm;
    int yi = i >> 5, xi = i & 31;
    for (int j = sub; j < 1024; j += 8) {
      float p = sP[r][j] * inv;
      sP[r][j] = p;
      int yj = j >> 5, xj = j & 31;
      int dy = min(max(yi - yj, -7), 7) + 7;
      int dx = min(max(xi - xj, -7), 7) + 7;
      atomicAdd(&sB[r][dy * 15 + dx], p);
    }
  }
  __syncthreads();

  // Phase 3: O = P.V (WMMA, one 16-wide d-tile per wave) + bsum.rpe_v
  {
    v8f c = {};
    int d = wv * 16 + (lane & 15);
    for (int kc = 0; kc < 32; ++kc) {
      Frag A, B;
#pragma unroll
      for (int j = 0; j < 8; ++j) {
        int k0 = kc * 32 + a_k0(j, lane);
        int m = lane & 15;
        A.u[j] = pack2bf(sP[m][k0], sP[m][k0 + 1]);
        B.u[j] = v32[(((size_t)bh * 64 + d) * 1024 + kc * 32 + b_k0(j, lane)) >> 1];
      }
      c = wmma_bf16(A, B, c);
    }
    // rpe_v epilogue: one table load per (k,d), shared by all 8 C elements
    float acc[8];
#pragma unroll
    for (int r = 0; r < 8; ++r) acc[r] = c[r];
    const int mbase = (lane < 16) ? 0 : 8;
    for (int kbkt = 0; kbkt < 225; ++kbkt) {
      float rv = rpe_v[((size_t)h * 225 + kbkt) * 64 + d];
#pragma unroll
      for (int r = 0; r < 8; ++r) acc[r] += sB[r + mbase][kbkt] * rv;
    }
#pragma unroll
    for (int r = 0; r < 8; ++r) {
      int tok = qtile * 16 + r + mbase;
      obf[((size_t)b * 1024 + tok) * 512 + h * 64 + d] = (__bf16)acc[r];
    }
  }
}

// ---------------- output projection + bias + [B,N,C] -> [B,C,N] transpose ----------------
__global__ void k_proj(const __bf16* __restrict__ a, const __bf16* __restrict__ wp,
                       const float* __restrict__ bias, float* __restrict__ out) {
  const int lane = threadIdx.x & 31;
  const int octile = blockIdx.x;  // 0..31
  const int mtile = blockIdx.y;   // 0..255
  const uint32_t* a32 = (const uint32_t*)a;
  const uint32_t* w32 = (const uint32_t*)wp;
  const int row = mtile * 16 + (lane & 15);
  const int oc_b = octile * 16 + (lane & 15);
  v8f c = {};
  for (int kc = 0; kc < 16; ++kc) {
    Frag A, B;
#pragma unroll
    for (int j = 0; j < 8; ++j) {
      A.u[j] = a32[((size_t)row * 512 + kc * 32 + a_k0(j, lane)) >> 1];
      B.u[j] = w32[((size_t)oc_b * 512 + kc * 32 + b_k0(j, lane)) >> 1];
    }
    c = wmma_bf16(A, B, c);
  }
  const int oc = octile * 16 + (lane & 15);
  const float bv = bias[oc];
#pragma unroll
  for (int r = 0; r < 8; ++r) {
    int m = r + ((lane < 16) ? 0 : 8);
    int tokg = mtile * 16 + m;
    int b = tokg >> 10, tok = tokg & 1023;
    out[((size_t)b * 512 + oc) * 1024 + tok] = c[r] + bv;
  }
}

// ---------------- workspace layout ----------------
static constexpr size_t SZ_QBF   = 32ull * 1024 * 64 * 2;     // 4 MiB each
static constexpr size_t OFF_QBF  = 0;
static constexpr size_t OFF_KBF  = OFF_QBF + SZ_QBF;
static constexpr size_t OFF_VBF  = OFF_KBF + SZ_QBF;
static constexpr size_t OFF_WQKV = OFF_VBF + SZ_QBF;                 // 1536*512*2
static constexpr size_t OFF_WPRJ = OFF_WQKV + 1536ull * 512 * 2;     // 512*512*2
static constexpr size_t OFF_RPEK = OFF_WPRJ + 512ull * 512 * 2;      // 8*225*64*2
static constexpr size_t OFF_RPEQ = OFF_RPEK + 8ull * 225 * 64 * 2;
static constexpr size_t OFF_TQ   = OFF_RPEQ + 8ull * 225 * 64 * 2;   // 32*1024*240*4
static constexpr size_t OFF_TK   = OFF_TQ + 32ull * 1024 * 240 * 4;
static constexpr size_t OFF_OBF  = OFF_TK + 32ull * 1024 * 240 * 4;  // 4096*512*2

extern "C" void kernel_launch(void* const* d_in, const int* in_sizes, int n_in,
                              void* d_out, int out_size, void* d_ws, size_t ws_size,
                              hipStream_t stream) {
  const float* x      = (const float*)d_in[0];
  const float* qkv_w  = (const float*)d_in[1];
  const float* proj_w = (const float*)d_in[2];
  const float* proj_b = (const float*)d_in[3];
  const float* rpe_q  = (const float*)d_in[4];
  const float* rpe_k  = (const float*)d_in[5];
  const float* rpe_v  = (const float*)d_in[6];
  float* out = (float*)d_out;

  char* w = (char*)d_ws;
  __bf16* qb   = (__bf16*)(w + OFF_QBF);
  __bf16* kb   = (__bf16*)(w + OFF_KBF);
  __bf16* vb   = (__bf16*)(w + OFF_VBF);
  __bf16* wqkv = (__bf16*)(w + OFF_WQKV);
  __bf16* wprj = (__bf16*)(w + OFF_WPRJ);
  __bf16* rpek = (__bf16*)(w + OFF_RPEK);
  __bf16* rpeq = (__bf16*)(w + OFF_RPEQ);
  float*  tq   = (float*)(w + OFF_TQ);
  float*  tk   = (float*)(w + OFF_TK);
  __bf16* obf  = (__bf16*)(w + OFF_OBF);

  const float scale = 0.125f;  // 64^-0.5

  k_cvt_bf16<<<(1536 * 512 + 255) / 256, 256, 0, stream>>>(qkv_w, wqkv, 1536 * 512, 1.f);
  k_cvt_bf16<<<(512 * 512 + 255) / 256, 256, 0, stream>>>(proj_w, wprj, 512 * 512, 1.f);
  k_cvt_bf16<<<(8 * 225 * 64 + 255) / 256, 256, 0, stream>>>(rpe_k, rpek, 8 * 225 * 64, 1.f);
  k_cvt_bf16<<<(8 * 225 * 64 + 255) / 256, 256, 0, stream>>>(rpe_q, rpeq, 8 * 225 * 64, scale);

  k_qkv<<<dim3(96, 256), 32, 0, stream>>>(x, wqkv, qb, kb, vb);

  k_rpe_gemm<<<dim3(15, 64, 32), 32, 0, stream>>>(qb, rpek, tq);  // tq = q_s . rpe_k^T
  k_rpe_gemm<<<dim3(15, 64, 32), 32, 0, stream>>>(kb, rpeq, tk);  // tk = k . (s*rpe_q)^T

  k_attn<<<dim3(64, 32), 128, 0, stream>>>(qb, kb, vb, tq, tk, rpe_v, obf);

  k_proj<<<dim3(32, 256), 32, 0, stream>>>(obf, wprj, proj_b, out);
}
